// GNNEncoder_14388140441820
// MI455X (gfx1250) — compile-verified, matrix operand order checked by reading, not run on hardware
//
#include <hip/hip_runtime.h>
#include <hip/hip_bf16.h>

typedef __attribute__((ext_vector_type(16))) _Float16 v16h;
typedef __attribute__((ext_vector_type(8)))  _Float16 v8h;
typedef __attribute__((ext_vector_type(8)))  float    v8f;

#define N_NODES 50000
#define N_EDGES 800000
#define C_IN    128
#define C_HID   64          // == C_OUT; feature width of both GEMM outputs
#define MTILES  (N_NODES / 16)   // 3125, exact
#define MT_PER_WAVE 8

// ---------------------------------------------------------------- degree
__global__ void k_deg_init(float* __restrict__ deg, int n) {
    int i = blockIdx.x * blockDim.x + threadIdx.x;
    if (i < n) deg[i] = 1.0f;               // self-loop contributes 1
}

__global__ void k_deg_edges(const long long* __restrict__ dst,
                            float* __restrict__ deg, int e) {
    int i = blockIdx.x * blockDim.x + threadIdx.x;
    if (i < e) {
        __hip_atomic_fetch_add(&deg[(int)dst[i]], 1.0f,
                               __ATOMIC_RELAXED, __HIP_MEMORY_SCOPE_AGENT);
    }
}

__global__ void k_dinv(float* __restrict__ deg, int n) {
    int i = blockIdx.x * blockDim.x + threadIdx.x;
    if (i < n) deg[i] = rsqrtf(deg[i]);     // deg >= 1 always (self-loops)
}

// ---------------------------------------------------------------- conversions
__global__ void k_cvt_f16(const float* __restrict__ in,
                          _Float16* __restrict__ out, int n) {
    int i = blockIdx.x * blockDim.x + threadIdx.x;
    if (i < n) out[i] = (_Float16)in[i];
}

// W [K][64] f32 (row-major)  ->  Wt [64][K] f16 (N-major) so B fragments are
// contiguous 16B vector loads per lane.
__global__ void k_cvt_w_t(const float* __restrict__ W,
                          _Float16* __restrict__ Wt, int K) {
    int idx = blockIdx.x * blockDim.x + threadIdx.x;
    if (idx < K * 64) {
        int k = idx >> 6;
        int n = idx & 63;
        Wt[n * K + k] = (_Float16)W[idx];
    }
}

__global__ void k_relu_cvt_f16(const float* __restrict__ in,
                               _Float16* __restrict__ out, int n) {
    int i = blockIdx.x * blockDim.x + threadIdx.x;
    if (i < n) out[i] = (_Float16)fmaxf(in[i], 0.0f);
}

// ---------------------------------------------------------------- WMMA GEMM
// C[M,64] = A[M,KD] (f16 row-major) x W[KD,64], with W given transposed as
// Bt[64][KD] f16. Fused epilogue: H = C (raw), AGG = bias + C * dinv^2
// (bias + self-loop term of the GCN aggregation).
// One wave32 owns one N-tile (of 4) and MT_PER_WAVE consecutive M-tiles,
// holding the full B fragment set for its ntile in registers. All A
// fragments for a tile are issued before the wmma chain so load latency
// hides under the accumulate chain.
template <int KD>
__global__ void k_gemm_wmma(const _Float16* __restrict__ A,
                            const _Float16* __restrict__ Bt,
                            const float* __restrict__ dinv,
                            const float* __restrict__ bias,
                            float* __restrict__ H,
                            float* __restrict__ AGG) {
    constexpr int KSTEPS = KD / 32;
    int wave   = blockIdx.x * (blockDim.x >> 5) + (threadIdx.x >> 5);
    int ntile  = wave & 3;
    int mgroup = wave >> 2;
    int m0     = mgroup * MT_PER_WAVE;
    if (m0 >= MTILES) return;               // wave-uniform: EXEC stays all-ones
    int m1 = (m0 + MT_PER_WAVE < MTILES) ? m0 + MT_PER_WAVE : MTILES;

    int lane   = threadIdx.x & 31;
    int l15    = lane & 15;
    int hihalf = lane >> 4;                 // 0: lanes 0-15, 1: lanes 16-31
    int koff   = hihalf << 3;               // K sub-offset 0 or 8 (ISA layout)
    int ncol   = ntile * 16 + l15;          // B/C column this lane feeds
    float bv   = bias[ncol];

    // B fragments for this ntile, all K-steps, held in registers.
    // 16-bit fragment layout (both A and B sides):
    //   lanes 0-15 : halves 0-7 = K ks*32+0..7,  halves 8-15 = K ks*32+16..23
    //   lanes 16-31: halves 0-7 = K ks*32+8..15, halves 8-15 = K ks*32+24..31
    v16h bf[KSTEPS];
#pragma unroll
    for (int ks = 0; ks < KSTEPS; ++ks) {
        const _Float16* bp = Bt + ncol * KD + ks * 32 + koff;
        v8h lo = *(const v8h*)(bp);
        v8h hi = *(const v8h*)(bp + 16);
#pragma unroll
        for (int i = 0; i < 8; ++i) { bf[ks][i] = lo[i]; bf[ks][8 + i] = hi[i]; }
    }

    for (int mt = m0; mt < m1; ++mt) {
        int mrow = mt * 16 + l15;
        const _Float16* arow = A + mrow * KD;
        if (mt + 1 < m1)                     // prefetch next tile's A rows
            __builtin_prefetch(arow + 16 * KD, 0, 0);

        // Phase 1: issue ALL A-fragment loads for this tile (independent).
        v16h af[KSTEPS];
#pragma unroll
        for (int ks = 0; ks < KSTEPS; ++ks) {
            const _Float16* ap = arow + ks * 32 + koff;
            v8h lo = *(const v8h*)(ap);
            v8h hi = *(const v8h*)(ap + 16);
#pragma unroll
            for (int i = 0; i < 8; ++i) { af[ks][i] = lo[i]; af[ks][8 + i] = hi[i]; }
        }

        // Phase 2: accumulate chain; later steps' waits overlap earlier wmmas.
        v8f acc = {};
#pragma unroll
        for (int ks = 0; ks < KSTEPS; ++ks) {
            acc = __builtin_amdgcn_wmma_f32_16x16x32_f16(
                false, af[ks], false, bf[ks], (short)0, acc, false, false);
        }

        // D f32 16x16 layout: VGPR v -> row v (lanes 0-15) or 8+v (lanes 16-31)
        int mbase = mt * 16 + (hihalf << 3);
#pragma unroll
        for (int v = 0; v < 8; ++v) {
            int row = mbase + v;
            int ci  = row * 64 + ncol;
            float val = acc[v];
            float di  = dinv[row];
            H[ci]   = val;
            AGG[ci] = bv + val * di * di;   // bias + self-loop term
        }
    }
}

// ---------------------------------------------------------------- edge scatter
// AGG[dst, c] += H[src, c] * dinv[src]*dinv[dst]; one thread = (edge, 4 chans)
__global__ void k_edge_agg(const float* __restrict__ H,
                           const float* __restrict__ dinv,
                           const long long* __restrict__ src,
                           const long long* __restrict__ dst,
                           float* __restrict__ AGG, int total) {
    int idx = blockIdx.x * blockDim.x + threadIdx.x;
    if (idx < total) {
        int e  = idx >> 4;                  // 16 threads per edge
        int c4 = (idx & 15) << 2;           // 4 channels each
        int s  = (int)src[e];
        int d  = (int)dst[e];
        float w = dinv[s] * dinv[d];
        const float4 hv = *(const float4*)(H + s * 64 + c4);
        float* op = AGG + d * 64 + c4;
        __hip_atomic_fetch_add(op + 0, hv.x * w, __ATOMIC_RELAXED, __HIP_MEMORY_SCOPE_AGENT);
        __hip_atomic_fetch_add(op + 1, hv.y * w, __ATOMIC_RELAXED, __HIP_MEMORY_SCOPE_AGENT);
        __hip_atomic_fetch_add(op + 2, hv.z * w, __ATOMIC_RELAXED, __HIP_MEMORY_SCOPE_AGENT);
        __hip_atomic_fetch_add(op + 3, hv.w * w, __ATOMIC_RELAXED, __HIP_MEMORY_SCOPE_AGENT);
    }
}

// ---------------------------------------------------------------- launch
extern "C" void kernel_launch(void* const* d_in, const int* in_sizes, int n_in,
                              void* d_out, int out_size, void* d_ws, size_t ws_size,
                              hipStream_t stream) {
    const float*     x   = (const float*)d_in[0];
    const long long* ei  = (const long long*)d_in[1];   // int64 [2, 800000]
    const float*     W1  = (const float*)d_in[2];
    const float*     b1  = (const float*)d_in[3];
    const float*     W2  = (const float*)d_in[4];
    const float*     b2  = (const float*)d_in[5];
    float*           out = (float*)d_out;

    const long long* srcv = ei;             // row 0
    const long long* dstv = ei + N_EDGES;   // row 1

    // workspace layout (256B-aligned slices), ~45 MB total
    char* ws = (char*)d_ws;
    float*    dinv = (float*)(ws);                       // 200,000 B
    _Float16* xh   = (_Float16*)(ws + 200192);           // 12.8 MB
    _Float16* w1t  = (_Float16*)(ws + 13000192);         // 16 KB (transposed)
    _Float16* w2t  = (_Float16*)(ws + 13016576);         // 8 KB (transposed)
    float*    hbuf = (float*)(ws + 13024768);            // 12.8 MB (h1 then h2)
    float*    agg  = (float*)(ws + 25824768);            // 12.8 MB
    _Float16* a1h  = (_Float16*)(ws + 38624768);         // 6.4 MB

    const int BS = 256;
    const int n_nc   = N_NODES * C_HID;                  // 3,200,000
    const int n_xin  = N_NODES * C_IN;                   // 6,400,000
    const int n_eagg = N_EDGES * 16;                     // 12,800,000 threads
    const int waves  = 4 * ((MTILES + MT_PER_WAVE - 1) / MT_PER_WAVE); // 1564
    const int gemm_blocks = (waves + 7) / 8;             // 8 waves / block

    // degree -> dinv
    k_deg_init <<<(N_NODES + BS - 1) / BS, BS, 0, stream>>>(dinv, N_NODES);
    k_deg_edges<<<(N_EDGES + BS - 1) / BS, BS, 0, stream>>>(dstv, dinv, N_EDGES);
    k_dinv     <<<(N_NODES + BS - 1) / BS, BS, 0, stream>>>(dinv, N_NODES);

    // f16 operands (weights transposed for vectorized B fragments)
    k_cvt_f16<<<(n_xin + BS - 1) / BS, BS, 0, stream>>>(x, xh, n_xin);
    k_cvt_w_t<<<(C_IN * 64 + BS - 1) / BS, BS, 0, stream>>>(W1, w1t, C_IN);
    k_cvt_w_t<<<(C_HID * 64 + BS - 1) / BS, BS, 0, stream>>>(W2, w2t, C_HID);

    // layer 1: h1 = x @ W1 (epilogue writes h1 and agg = b1 + h1*dinv^2)
    k_gemm_wmma<C_IN><<<gemm_blocks, BS, 0, stream>>>(xh, w1t, dinv, b1, hbuf, agg);
    k_edge_agg <<<(n_eagg + BS - 1) / BS, BS, 0, stream>>>(hbuf, dinv, srcv, dstv, agg, n_eagg);
    k_relu_cvt_f16<<<(n_nc + BS - 1) / BS, BS, 0, stream>>>(agg, a1h, n_nc);

    // layer 2: h2 = relu(h1') @ W2 (epilogue writes h2 and d_out = b2 + h2*dinv^2)
    k_gemm_wmma<C_HID><<<gemm_blocks, BS, 0, stream>>>(a1h, w2t, dinv, b2, hbuf, out);
    k_edge_agg <<<(n_eagg + BS - 1) / BS, BS, 0, stream>>>(hbuf, dinv, srcv, dstv, out, n_eagg);
}
